// RPN_82772609728802
// MI455X (gfx1250) — matrix operand order repaired and checked by reading, not compiled, not run on hardware
//
#include <hip/hip_runtime.h>
#include <hip/hip_bf16.h>
#include <math.h>

typedef __attribute__((ext_vector_type(16))) __bf16 v16bf;
typedef __attribute__((ext_vector_type(8)))  float  v8f;

#if __has_builtin(__builtin_amdgcn_global_load_async_to_lds_b128) && \
    __has_builtin(__builtin_amdgcn_s_wait_asynccnt)
#define HAVE_ASYNC 1
#else
#define HAVE_ASYNC 0
#endif

// async b128 builtin operand types: 16B int vector pointers, global (AS1) / LDS (AS3)
typedef int v4i_b128 __attribute__((vector_size(16)));
typedef __attribute__((address_space(1))) v4i_b128* async_gptr;
typedef __attribute__((address_space(3))) v4i_b128* async_lptr;

#define N_IMG   4
#define C_INCH  256
#define C_OUTCH 256
#define H_DIM   128
#define W_DIM   128
#define A_ANCH  9
#define NPIX    (H_DIM * W_DIM)        // 16384
#define NLOG    (NPIX * A_ANCH)        // 147456
#define PRE_K   1000
#define POST_K  1000
#define NMS_THR 0.7f
#define SCL_CLMP 4.135166556742356f    // log(1000/16)
#define NEGV    (-1000000000.0f)
#define IMG_SZ  1024.0f

#define LDA 48   // LDS leading dim (elements): keeps every 16B fragment load 16B/32B aligned
#define LDB 48

static __device__ __forceinline__ int gtid() { return blockIdx.x * blockDim.x + threadIdx.x; }
static __device__ __forceinline__ int gstride() { return gridDim.x * blockDim.x; }

// stage one 16B slice global->LDS (async on CDNA5; sync fallback otherwise)
static __device__ __forceinline__ void stage16(const __bf16* g, __bf16* l, bool oob) {
#if HAVE_ASYNC
    if (!oob) {
        __builtin_amdgcn_global_load_async_to_lds_b128((async_gptr)g, (async_lptr)l, 0, 0);
    } else {
        const uint4 zz = {0u, 0u, 0u, 0u};
        *(uint4*)l = zz;
    }
#else
    uint4 v = {0u, 0u, 0u, 0u};
    if (!oob) v = *(const uint4*)g;
    *(uint4*)l = v;
#endif
}

static __device__ __forceinline__ void wait_cur_chunk(bool more) {
#if HAVE_ASYNC
    if (more) __builtin_amdgcn_s_wait_asynccnt(3);   // 3 newer ops (next chunk) may remain
    else      __builtin_amdgcn_s_wait_asynccnt(0);
#else
    (void)more;
#endif
}

// ---------------------------------------------------------------- zero scratch
__global__ void k_zero(unsigned* p, int n) {
    for (int i = gtid(); i < n; i += gstride()) p[i] = 0u;
}

// ------------------------------------------------- pack 3x3 weights -> bf16 [tap][co][ci]
__global__ void k_pack_w(const float* __restrict__ w, __bf16* __restrict__ wpk) {
    const int total = 9 * C_OUTCH * C_INCH;
    for (int i = gtid(); i < total; i += gstride()) {
        int ci  = i % C_INCH;
        int co  = (i / C_INCH) % C_OUTCH;
        int tap = i / (C_INCH * C_OUTCH);
        wpk[i] = (__bf16)w[(co * C_INCH + ci) * 9 + tap];
    }
}

// --------------------------------- pack head weights: rows 0..35 bbox, 36..44 cls, pad 64
__global__ void k_pack_h(const float* __restrict__ bw, const float* __restrict__ bb,
                         const float* __restrict__ cw, const float* __restrict__ cb,
                         __bf16* __restrict__ hpk, float* __restrict__ hbias) {
    const int total = 64 * C_INCH;
    for (int i = gtid(); i < total; i += gstride()) {
        int ci = i % C_INCH, row = i / C_INCH;
        float v = 0.0f;
        if (row < 36)       v = bw[row * C_INCH + ci];
        else if (row < 45)  v = cw[(row - 36) * C_INCH + ci];
        hpk[i] = (__bf16)v;
        if (i < 64) hbias[i] = (i < 36) ? bb[i] : ((i < 45) ? cb[i - 36] : 0.0f);
    }
}

// ------------------------------------------------- NCHW f32 -> NHWC bf16
__global__ void k_cvt(const float* __restrict__ f, __bf16* __restrict__ o) {
    const long total = (long)N_IMG * C_INCH * H_DIM * W_DIM;
    for (long i = gtid(); i < total; i += gstride()) {
        int c = (int)(i & 255);
        long t = i >> 8;
        int x = (int)(t & 127);
        int y = (int)((t >> 7) & 127);
        int n = (int)(t >> 14);
        o[i] = (__bf16)f[(((long)n * C_INCH + c) * H_DIM + y) * W_DIM + x];
    }
}

union Frag { uint4 q[2]; v16bf v; };

// ---------------------------------------------------------------- 3x3 conv via WMMA
// block = (n, y, cout_block_of_64); M=128 pixels (x), N=64 couts, K = taps * 256 cin.
// Double-buffered LDS; async global->LDS staging overlapped with WMMA on previous chunk.
__global__ __launch_bounds__(256) void k_conv3(const __bf16* __restrict__ fb,
                                               const __bf16* __restrict__ wpk,
                                               const float*  __restrict__ bias,
                                               __bf16* __restrict__ xb) {
    __shared__ __bf16 sA[2][128 * LDA];
    __shared__ __bf16 sB[2][64 * LDB];
    __shared__ int s_tap[10];          // [0..8] packed (kx | tap<<4 | yy<<16), [9]=count

    const int bid = blockIdx.x;
    const int cb  = bid & 3;
    const int y   = (bid >> 2) & 127;
    const int n   = bid >> 9;

    const int tid  = threadIdx.x;
    const int lane = tid & 31;
    const int wid  = tid >> 5;
    const int waveM = wid & 3;     // 4 waves along M (32 rows each)
    const int waveN = wid >> 2;    // 2 waves along N (32 cols each)

    if (tid == 0) {
        int cnt = 0;
#pragma unroll
        for (int t = 0; t < 9; ++t) {
            const int ky = t / 3, kx = t % 3;     // compile-time
            const int yy = y + ky - 1;
            if (yy >= 0 && yy < H_DIM) s_tap[cnt++] = kx | (t << 4) | (yy << 16);
        }
        s_tap[9] = cnt;
    }

    const v8f z = {0.f,0.f,0.f,0.f,0.f,0.f,0.f,0.f};
    v8f acc[2][2] = {{z, z}, {z, z}};

    const int ar = tid >> 1, ah = tid & 1;   // A staging: pixel row 0..127, 16-elem half
    const int br = tid >> 2, bq = tid & 3;   // B staging: cout row 0..63, 8-elem quarter

    __syncthreads();
    const int total = s_tap[9] * 8;          // (valid taps) x (8 chunks of 32 cin)

    auto issue = [&](int c, int p) {
        const int info = s_tap[c >> 3];
        const int kc   = c & 7;
        const int kx   = info & 15;
        const int tap  = (info >> 4) & 15;
        const int yy   = info >> 16;
        const int xs   = ar + kx - 1;
        const bool oob = (xs < 0) || (xs >= W_DIM);
        const __bf16* ga = fb + (((long)n * H_DIM + yy) * W_DIM + xs) * C_INCH + ah * 16 + kc * 32;
        const __bf16* gb = wpk + ((long)(tap * C_OUTCH + cb * 64 + br)) * C_INCH + bq * 8 + kc * 32;
        __bf16* la = &sA[p][ar * LDA + ah * 16];
        __bf16* lb = &sB[p][br * LDB + bq * 8];
        stage16(ga,     la,     oob);
        stage16(ga + 8, la + 8, oob);
        stage16(gb,     lb,     false);
        if (!oob) __builtin_prefetch((const void*)(ga + 32), 0, 1);   // next chunk into L2
    };

    issue(0, 0);

    for (int c = 0; c < total; ++c) {
        const int p = c & 1;
        const bool more = (c + 1 < total);
        if (more) issue(c + 1, p ^ 1);
        wait_cur_chunk(more);
        __syncthreads();

        const int ksel = lane >> 4;
        Frag fa[2], fbm[2];
#pragma unroll
        for (int tm = 0; tm < 2; ++tm) {
            int r = waveM * 32 + tm * 16 + (lane & 15);
            fa[tm].q[0] = *(const uint4*)(&sA[p][r * LDA + ksel * 16]);
            fa[tm].q[1] = *(const uint4*)(&sA[p][r * LDA + ksel * 16 + 8]);
        }
#pragma unroll
        for (int tn = 0; tn < 2; ++tn) {
            int cc = waveN * 32 + tn * 16 + (lane & 15);
            fbm[tn].q[0] = *(const uint4*)(&sB[p][cc * LDB + ksel * 16]);
            fbm[tn].q[1] = *(const uint4*)(&sB[p][cc * LDB + ksel * 16 + 8]);
        }
#pragma unroll
        for (int tm = 0; tm < 2; ++tm)
#pragma unroll
            for (int tn = 0; tn < 2; ++tn)
                acc[tm][tn] = __builtin_amdgcn_wmma_f32_16x16x32_bf16(
                    false, fa[tm].v, false, fbm[tn].v, (short)0, acc[tm][tn], false, false);
        __syncthreads();
    }

    // epilogue: bias + ReLU, write NHWC bf16
#pragma unroll
    for (int tm = 0; tm < 2; ++tm)
#pragma unroll
        for (int tn = 0; tn < 2; ++tn) {
            const int c   = cb * 64 + waveN * 32 + tn * 16 + (lane & 15);
            const float bv = bias[c];
            const int mrow = waveM * 32 + tm * 16 + ((lane >> 4) << 3);
#pragma unroll
            for (int r = 0; r < 8; ++r) {
                float v = acc[tm][tn][r] + bv;
                v = v > 0.f ? v : 0.f;
                const int x = mrow + r;
                xb[(((long)n * H_DIM + y) * W_DIM + x) * C_OUTCH + c] = (__bf16)v;
            }
        }
}

// ---------------------------------------------------------------- 1x1 heads via WMMA
// block = (n, y); M=128 pixels, N=64 (45 live), K=256; same double-buffered pipeline
__global__ __launch_bounds__(256) void k_head(const __bf16* __restrict__ xb,
                                              const __bf16* __restrict__ hpk,
                                              const float*  __restrict__ hbias,
                                              float* __restrict__ logits,
                                              float* __restrict__ deltas) {
    __shared__ __bf16 sA[2][128 * LDA];
    __shared__ __bf16 sB[2][64 * LDB];

    const int bid = blockIdx.x;
    const int y = bid & 127;
    const int n = bid >> 7;

    const int tid  = threadIdx.x;
    const int lane = tid & 31;
    const int wid  = tid >> 5;
    const int waveM = wid & 3;
    const int waveN = wid >> 2;

    const v8f z = {0.f,0.f,0.f,0.f,0.f,0.f,0.f,0.f};
    v8f acc[2][2] = {{z, z}, {z, z}};

    const int ar = tid >> 1, ah = tid & 1;
    const int br = tid >> 2, bq = tid & 3;

    auto issue = [&](int kc, int p) {
        const __bf16* ga = xb + (((long)n * H_DIM + y) * W_DIM + ar) * C_OUTCH + ah * 16 + kc * 32;
        const __bf16* gb = hpk + (long)br * C_INCH + bq * 8 + kc * 32;
        stage16(ga,     &sA[p][ar * LDA + ah * 16],     false);
        stage16(ga + 8, &sA[p][ar * LDA + ah * 16 + 8], false);
        stage16(gb,     &sB[p][br * LDB + bq * 8],      false);
    };

    issue(0, 0);

    for (int kc = 0; kc < 8; ++kc) {
        const int p = kc & 1;
        const bool more = (kc + 1 < 8);
        if (more) issue(kc + 1, p ^ 1);
        wait_cur_chunk(more);
        __syncthreads();

        const int ksel = lane >> 4;
        Frag fa[2], fbm[2];
#pragma unroll
        for (int tm = 0; tm < 2; ++tm) {
            int r = waveM * 32 + tm * 16 + (lane & 15);
            fa[tm].q[0] = *(const uint4*)(&sA[p][r * LDA + ksel * 16]);
            fa[tm].q[1] = *(const uint4*)(&sA[p][r * LDA + ksel * 16 + 8]);
        }
#pragma unroll
        for (int tn = 0; tn < 2; ++tn) {
            int cc = waveN * 32 + tn * 16 + (lane & 15);
            fbm[tn].q[0] = *(const uint4*)(&sB[p][cc * LDB + ksel * 16]);
            fbm[tn].q[1] = *(const uint4*)(&sB[p][cc * LDB + ksel * 16 + 8]);
        }
#pragma unroll
        for (int tm = 0; tm < 2; ++tm)
#pragma unroll
            for (int tn = 0; tn < 2; ++tn)
                acc[tm][tn] = __builtin_amdgcn_wmma_f32_16x16x32_bf16(
                    false, fa[tm].v, false, fbm[tn].v, (short)0, acc[tm][tn], false, false);
        __syncthreads();
    }

    // scatter into reference layouts:
    // deltas[n][(pix*9 + a)*4 + comp], logits[n][pix*9 + a], with c = a*4+comp (c<36) / c-36 (36<=c<45)
#pragma unroll
    for (int tm = 0; tm < 2; ++tm)
#pragma unroll
        for (int tn = 0; tn < 2; ++tn) {
            const int c   = waveN * 32 + tn * 16 + (lane & 15);
            const float bv = hbias[c];
            const int mrow = waveM * 32 + tm * 16 + ((lane >> 4) << 3);
#pragma unroll
            for (int r = 0; r < 8; ++r) {
                const float v = acc[tm][tn][r] + bv;
                const int x   = mrow + r;
                const int pix = y * W_DIM + x;
                if (c < 36) {
                    const int a = c >> 2, comp = c & 3;
                    deltas[((long)n * NLOG + (long)pix * A_ANCH + a) * 4 + comp] = v;
                } else if (c < 45) {
                    logits[(long)n * NLOG + (long)pix * A_ANCH + (c - 36)] = v;
                }
            }
        }
}

// ---------------------------------------------------------------- top-k selection
static __device__ __forceinline__ unsigned fkey(float f) {
    unsigned u = __float_as_uint(f);
    return (u & 0x80000000u) ? ~u : (u | 0x80000000u);
}

__global__ void k_hist(const float* __restrict__ logits, unsigned* __restrict__ hist) {
    const int total = N_IMG * NLOG;
    for (int i = gtid(); i < total; i += gstride()) {
        const int n = i / NLOG;
        atomicAdd(&hist[n * 2048 + (fkey(logits[i]) >> 21)], 1u);
    }
}

// meta: [0..3]=threshold bin, [4..7]=count strictly above, [8..11]=cnt_hi, [12..15]=cnt_eq
__global__ void k_thresh(const unsigned* __restrict__ hist, unsigned* __restrict__ meta) {
    if (threadIdx.x != 0) return;
    const int n = blockIdx.x;
    const unsigned* h = hist + n * 2048;
    unsigned acc = 0, above = 0; int T = 0;
    for (int b = 2047; b >= 0; --b) {
        unsigned c = h[b];
        if (acc + c >= PRE_K) { T = b; above = acc; break; }
        acc += c;
    }
    meta[n] = (unsigned)T;
    meta[4 + n] = above;
}

__global__ void k_compact(const float* __restrict__ logits, unsigned* __restrict__ meta,
                          int* __restrict__ sel) {
    const int total = N_IMG * NLOG;
    for (int i = gtid(); i < total; i += gstride()) {
        const int n = i / NLOG;
        const int li = i - n * NLOG;
        const unsigned bin = fkey(logits[i]) >> 21;
        const unsigned T = meta[n];
        if (bin > T) {
            unsigned p = atomicAdd(&meta[8 + n], 1u);
            if (p < PRE_K) sel[n * 1024 + p] = li;
        } else if (bin == T) {
            unsigned p = meta[4 + n] + atomicAdd(&meta[12 + n], 1u);
            if (p < PRE_K) sel[n * 1024 + p] = li;
        }
    }
}

// ---------------------------------------------------------------- box decode
__global__ void k_decode(const float* __restrict__ logits, const float* __restrict__ deltas,
                         const int* __restrict__ sel,
                         float* __restrict__ boxsel, float* __restrict__ scsel) {
    const int n = blockIdx.x >> 2;
    const int j = (blockIdx.x & 3) * 256 + threadIdx.x;
    if (j >= PRE_K) return;
    const int li = sel[n * 1024 + j];
    const float score = logits[(long)n * NLOG + li];
    const int a = li % A_ANCH;
    const int pix = li / A_ANCH;
    const int x = pix % W_DIM, yq = pix / W_DIM;

    const float SZ[3] = {128.f, 256.f, 512.f};
    const float RT[3] = {0.5f, 1.f, 2.f};
    const float s = SZ[a / 3], r = RT[a % 3];
    const float bw = sqrtf(s * s / r), bh = bw * r;
    const float cx = (float)x * 8.f, cy = (float)yq * 8.f;

    const float* d = deltas + ((long)n * NLOG + li) * 4;
    const float dx = d[0], dy = d[1];
    const float dw = fminf(d[2], SCL_CLMP), dh = fminf(d[3], SCL_CLMP);
    const float pcx = dx * bw + cx, pcy = dy * bh + cy;
    const float pw = __expf(dw) * bw, ph = __expf(dh) * bh;

    float* o = boxsel + ((long)n * PRE_K + j) * 4;
    o[0] = fminf(fmaxf(pcx - 0.5f * pw, 0.f), IMG_SZ);
    o[1] = fminf(fmaxf(pcy - 0.5f * ph, 0.f), IMG_SZ);
    o[2] = fminf(fmaxf(pcx + 0.5f * pw, 0.f), IMG_SZ);
    o[3] = fminf(fmaxf(pcy + 0.5f * ph, 0.f), IMG_SZ);
    scsel[n * PRE_K + j] = score;
}

// ---------------------------------------------------------------- greedy NMS (1 block / image)
__global__ __launch_bounds__(256) void k_nms(const float* __restrict__ boxsel,
                                             const float* __restrict__ scsel,
                                             float* __restrict__ out) {
    __shared__ float bx[PRE_K * 4];
    __shared__ float sup[PRE_K];
    __shared__ float rv[256];
    __shared__ int   ri[256];
    __shared__ float cur[4];

    const int n = blockIdx.x, tid = threadIdx.x;
    for (int j = tid; j < PRE_K; j += 256) {
        sup[j] = scsel[n * PRE_K + j];
#pragma unroll
        for (int k = 0; k < 4; ++k) bx[j * 4 + k] = boxsel[((long)n * PRE_K + j) * 4 + k];
    }
    __syncthreads();

    for (int it = 0; it < POST_K; ++it) {
        float best = -INFINITY; int bi = 0;
        for (int j = tid; j < PRE_K; j += 256) {
            const float v = sup[j];
            if (v > best) { best = v; bi = j; }
        }
        rv[tid] = best; ri[tid] = bi;
        __syncthreads();
        for (int s = 128; s > 0; s >>= 1) {
            if (tid < s && rv[tid + s] > rv[tid]) { rv[tid] = rv[tid + s]; ri[tid] = ri[tid + s]; }
            __syncthreads();
        }
        if (tid == 0) {
            const int i = ri[0];
            const bool valid = rv[0] > NEGV * 0.5f;
            float* o = out + ((long)n * POST_K + it) * 5;
            if (valid) {
                o[0] = bx[i*4]; o[1] = bx[i*4+1]; o[2] = bx[i*4+2]; o[3] = bx[i*4+3]; o[4] = rv[0];
            } else {
                o[0] = o[1] = o[2] = o[3] = 0.f; o[4] = NEGV;
            }
            cur[0] = bx[i*4]; cur[1] = bx[i*4+1]; cur[2] = bx[i*4+2]; cur[3] = bx[i*4+3];
            sup[i] = NEGV;
        }
        __syncthreads();
        const float X1 = cur[0], Y1 = cur[1], X2 = cur[2], Y2 = cur[3];
        const float a1 = (X2 - X1) * (Y2 - Y1);
        for (int j = tid; j < PRE_K; j += 256) {
            const float jx1 = bx[j*4], jy1 = bx[j*4+1], jx2 = bx[j*4+2], jy2 = bx[j*4+3];
            const float ix = fmaxf(fminf(X2, jx2) - fmaxf(X1, jx1), 0.f);
            const float iy = fmaxf(fminf(Y2, jy2) - fmaxf(Y1, jy1), 0.f);
            const float inter = ix * iy;
            const float a2 = (jx2 - jx1) * (jy2 - jy1);
            if (inter / (a1 + a2 - inter + 1e-9f) > NMS_THR) sup[j] = NEGV;
        }
        __syncthreads();
    }
}

// ---------------------------------------------------------------- launch
extern "C" void kernel_launch(void* const* d_in, const int* in_sizes, int n_in,
                              void* d_out, int out_size, void* d_ws, size_t ws_size,
                              hipStream_t stream) {
    const float* features = (const float*)d_in[0];
    const float* conv1_w  = (const float*)d_in[1];
    const float* conv1_b  = (const float*)d_in[2];
    const float* bbox_w   = (const float*)d_in[3];
    const float* bbox_b   = (const float*)d_in[4];
    const float* cls_w    = (const float*)d_in[5];
    const float* cls_b    = (const float*)d_in[6];
    float* out = (float*)d_out;

    char* ws = (char*)d_ws;
    size_t off = 0;
    auto carve = [&](size_t bytes) -> char* {
        char* p = ws + off;
        off = (off + bytes + 255) & ~(size_t)255;
        return p;
    };

    __bf16* featB  = (__bf16*)carve(sizeof(__bf16) * (size_t)N_IMG * H_DIM * W_DIM * C_INCH);
    __bf16* xB     = (__bf16*)carve(sizeof(__bf16) * (size_t)N_IMG * H_DIM * W_DIM * C_OUTCH);
    __bf16* wpk    = (__bf16*)carve(sizeof(__bf16) * 9 * C_OUTCH * C_INCH);
    __bf16* hpk    = (__bf16*)carve(sizeof(__bf16) * 64 * C_INCH);
    float*  hbias  = (float*) carve(sizeof(float) * 64);
    float*  logits = (float*) carve(sizeof(float) * (size_t)N_IMG * NLOG);
    float*  deltas = (float*) carve(sizeof(float) * (size_t)N_IMG * NLOG * 4);
    unsigned* hist = (unsigned*)carve(sizeof(unsigned) * N_IMG * 2048);
    unsigned* meta = (unsigned*)carve(sizeof(unsigned) * 16);
    int*    sel    = (int*)   carve(sizeof(int) * N_IMG * 1024);
    float*  boxsel = (float*) carve(sizeof(float) * N_IMG * PRE_K * 4);
    float*  scsel  = (float*) carve(sizeof(float) * N_IMG * PRE_K);

    k_zero   <<<64,   256, 0, stream>>>(hist, N_IMG * 2048);
    k_zero   <<<1,    64,  0, stream>>>(meta, 16);
    k_pack_w <<<1152, 256, 0, stream>>>(conv1_w, wpk);
    k_pack_h <<<64,   256, 0, stream>>>(bbox_w, bbox_b, cls_w, cls_b, hpk, hbias);
    k_cvt    <<<4096, 256, 0, stream>>>(features, featB);
    k_conv3  <<<2048, 256, 0, stream>>>(featB, wpk, conv1_b, xB);
    k_head   <<<512,  256, 0, stream>>>(xB, hpk, hbias, logits, deltas);
    k_hist   <<<1024, 256, 0, stream>>>(logits, hist);
    k_thresh <<<4,    32,  0, stream>>>(hist, meta);
    k_compact<<<1024, 256, 0, stream>>>(logits, meta, sel);
    k_decode <<<16,   256, 0, stream>>>(logits, deltas, sel, boxsel, scsel);
    k_nms    <<<4,    256, 0, stream>>>(boxsel, scsel, out);
}
